// Contrastive_model_edges_features_47278999994907
// MI455X (gfx1250) — matrix operand order repaired and checked by reading, not compiled, not run on hardware
//
#include <hip/hip_runtime.h>

typedef __attribute__((ext_vector_type(16))) _Float16 v16h;
typedef __attribute__((ext_vector_type(8)))  _Float16 v8h;
typedef __attribute__((ext_vector_type(8)))  float    v8f;
typedef __attribute__((ext_vector_type(4)))  float    v4f;

// ---------------------------------------------------------------------------
// WMMA helper: one 16x16 output tile, K accumulated in steps of 32 (f16 A/B,
// f32 C). A comes from an LDS tile [16][ldk] (row-major f16), B from global
// pre-transposed weights Wt[N][K] (row-major f16). Contiguous k order used
// for A and B: A lane(0-15): K = k0+0..15 ; lane(16-31): K = k0+16..31.
//        B lane(0-15): K = k0+0..15 ; lane(16-31): K = k0+16..31, N = lane%16.
// ---------------------------------------------------------------------------
template<int K>
__device__ inline v8f gemm_tile(const _Float16* tIn, const _Float16* __restrict__ Wt,
                                int r, int hl, int nrow, float bz)
{
    v8f c = { bz, bz, bz, bz, bz, bz, bz, bz };
#pragma unroll
    for (int kt = 0; kt < K / 32; ++kt) {
        const _Float16* pa = tIn + r * K + kt * 32 + hl * 16;
        v16h a = *(const v16h*)(pa);
        v16h b = *(const v16h*)(Wt + (size_t)nrow * K + kt * 32 + hl * 16);
        c = __builtin_amdgcn_wmma_f32_16x16x32_f16(false, a, false, b, (short)0, c, false, false);
    }
    return c;
}

// ---------------------------------------------------------------------------
// Fused 3-layer edge MLP (128->128->128->128) + LayerNorm + atomic scatter
// of the segment_sum over dst. One wave = 16 edges x 128 features; all
// intermediates live in a wave-private LDS ping-pong tile (f16).
// ---------------------------------------------------------------------------
__global__ __launch_bounds__(256, 1)
void edge_mlp_scatter(const float* __restrict__ m,
                      const _Float16* __restrict__ W1t, const float* __restrict__ b1,
                      const _Float16* __restrict__ W2t, const float* __restrict__ b2,
                      const _Float16* __restrict__ W3t, const float* __restrict__ b3,
                      const float* __restrict__ gamma, const float* __restrict__ beta,
                      const int* __restrict__ dst, float* __restrict__ hout, long long E)
{
    __shared__ _Float16 smem[8 * 2 * 16 * 128];   // 64 KB: 8 waves x (A,B) ping-pong
    const int wave = threadIdx.x >> 5;
    const int lane = threadIdx.x & 31;
    const int r  = lane & 15;
    const int hl = lane >> 4;
    _Float16* tA = smem + wave * 4096;
    _Float16* tB = tA + 2048;

    const long long e0 = ((long long)blockIdx.x * 8 + wave) * 16;
    if (e0 >= E) return;

    // ---- stage 16x128 fp32 edge rows (contiguous 8 KB) -> f16 LDS tile ----
    const float* mp = m + e0 * 128;
    if (e0 + 16 <= E) {
        __builtin_prefetch(mp + 16 * 128, 0, 0);   // next tile -> global_prefetch
#pragma unroll
        for (int i = 0; i < 16; ++i) {
            int idx = (i * 32 + lane) * 4;
            v4f t = *(const v4f*)(mp + idx);
            tA[idx + 0] = (_Float16)t[0]; tA[idx + 1] = (_Float16)t[1];
            tA[idx + 2] = (_Float16)t[2]; tA[idx + 3] = (_Float16)t[3];
        }
    } else {
#pragma unroll
        for (int i = 0; i < 16; ++i) {
            int idx = (i * 32 + lane) * 4;
            long long e = e0 + (idx >> 7);
            if (e < E) {
                v4f t = *(const v4f*)(mp + idx);
                tA[idx + 0] = (_Float16)t[0]; tA[idx + 1] = (_Float16)t[1];
                tA[idx + 2] = (_Float16)t[2]; tA[idx + 3] = (_Float16)t[3];
            } else {
                tA[idx + 0] = (_Float16)0.f; tA[idx + 1] = (_Float16)0.f;
                tA[idx + 2] = (_Float16)0.f; tA[idx + 3] = (_Float16)0.f;
            }
        }
    }

    // ---- layer 1: tA -> relu -> tB ----
#pragma unroll
    for (int nt = 0; nt < 8; ++nt) {
        v8f c = gemm_tile<128>(tA, W1t, r, hl, nt * 16 + r, b1[nt * 16 + r]);
#pragma unroll
        for (int vv = 0; vv < 8; ++vv) {
            float x = c[vv]; x = x > 0.f ? x : 0.f;
            tB[(hl * 8 + vv) * 128 + nt * 16 + r] = (_Float16)x;
        }
    }
    // ---- layer 2: tB -> relu -> tA ----
#pragma unroll
    for (int nt = 0; nt < 8; ++nt) {
        v8f c = gemm_tile<128>(tB, W2t, r, hl, nt * 16 + r, b2[nt * 16 + r]);
#pragma unroll
        for (int vv = 0; vv < 8; ++vv) {
            float x = c[vv]; x = x > 0.f ? x : 0.f;
            tA[(hl * 8 + vv) * 128 + nt * 16 + r] = (_Float16)x;
        }
    }
    // ---- layer 3: keep f32 accumulators in registers for LayerNorm ----
    v8f a3[8];
#pragma unroll
    for (int nt = 0; nt < 8; ++nt)
        a3[nt] = gemm_tile<128>(tA, W3t, r, hl, nt * 16 + r, b3[nt * 16 + r]);

    // ---- LayerNorm over 128 features per row (rows live per 16-lane half) ----
    float s[8], q[8];
#pragma unroll
    for (int vv = 0; vv < 8; ++vv) { s[vv] = 0.f; q[vv] = 0.f; }
#pragma unroll
    for (int nt = 0; nt < 8; ++nt)
#pragma unroll
        for (int vv = 0; vv < 8; ++vv) { float x = a3[nt][vv]; s[vv] += x; q[vv] += x * x; }
#pragma unroll
    for (int mask = 1; mask < 16; mask <<= 1)
#pragma unroll
        for (int vv = 0; vv < 8; ++vv) {
            s[vv] += __shfl_xor(s[vv], mask, 32);
            q[vv] += __shfl_xor(q[vv], mask, 32);
        }
    float mu[8], rsg[8];
#pragma unroll
    for (int vv = 0; vv < 8; ++vv) {
        mu[vv] = s[vv] * 0.0078125f;
        float var = q[vv] * 0.0078125f - mu[vv] * mu[vv];
        rsg[vv] = rsqrtf(var + 1e-5f);
    }

    // ---- fused segment_sum: atomic scatter into h[dst] ----
    int dr[8];
#pragma unroll
    for (int vv = 0; vv < 8; ++vv) {
        long long e = e0 + hl * 8 + vv;
        dr[vv] = (e < E) ? dst[e] : -1;
    }
#pragma unroll
    for (int nt = 0; nt < 8; ++nt) {
        float g  = gamma[nt * 16 + r];
        float be = beta[nt * 16 + r];
#pragma unroll
        for (int vv = 0; vv < 8; ++vv) {
            if (dr[vv] >= 0) {
                float x = (a3[nt][vv] - mu[vv]) * rsg[vv] * g + be;
                atomicAdd(hout + (size_t)dr[vv] * 128 + nt * 16 + r, x);
            }
        }
    }
}

// ---------------------------------------------------------------------------
// Node-side GEMM: Y[N,Dout] = relu( (X * rsqrt(deg_in)) @ W + b ), WMMA,
// templated on K (128 / 256). grid.y picks a 128-wide column group.
// ---------------------------------------------------------------------------
template<int K>
__global__ __launch_bounds__(256, 1)
void node_gemm(const float* __restrict__ X, const float* __restrict__ rsIn,
               const _Float16* __restrict__ Wt, const float* __restrict__ bias,
               float* __restrict__ Y, long long N, int Dout)
{
    __shared__ _Float16 smem[8 * 16 * K];
    const int wave = threadIdx.x >> 5;
    const int lane = threadIdx.x & 31;
    const int r  = lane & 15;
    const int hl = lane >> 4;
    _Float16* tA = smem + wave * 16 * K;

    const long long n0 = ((long long)blockIdx.x * 8 + wave) * 16;
    if (n0 >= N) return;
    const int colBase = blockIdx.y * 128;

    // stage 16 node rows, pre-scaled by rsqrt(deg_in), as f16
#pragma unroll
    for (int row = 0; row < 16; ++row) {
        long long n = n0 + row;
        float sc = (n < N) ? rsIn[n] : 0.f;
        const float* xp = X + n * (long long)K;
        for (int c = lane * 4; c < K; c += 128) {
            if (n < N) {
                v4f t = *(const v4f*)(xp + c);
                tA[row * K + c + 0] = (_Float16)(t[0] * sc);
                tA[row * K + c + 1] = (_Float16)(t[1] * sc);
                tA[row * K + c + 2] = (_Float16)(t[2] * sc);
                tA[row * K + c + 3] = (_Float16)(t[3] * sc);
            } else {
                tA[row * K + c + 0] = (_Float16)0.f; tA[row * K + c + 1] = (_Float16)0.f;
                tA[row * K + c + 2] = (_Float16)0.f; tA[row * K + c + 3] = (_Float16)0.f;
            }
        }
    }

#pragma unroll
    for (int nt = 0; nt < 8; ++nt) {
        int col = colBase + nt * 16 + r;
        v8f c = gemm_tile<K>(tA, Wt, r, hl, col, bias[col]);
#pragma unroll
        for (int vv = 0; vv < 8; ++vv) {
            long long n = n0 + hl * 8 + vv;
            if (n < N) {
                float x = c[vv];
                Y[n * (long long)Dout + col] = x > 0.f ? x : 0.f;
            }
        }
    }
}

// ---------------------------------------------------------------------------
// GraphConv gather/scatter: agg[dst] += X[src] * rsqrt(deg_out[src]).
// One thread per (edge, 4 columns); shift = log2(K/4).
// ---------------------------------------------------------------------------
__global__ void edge_scatter(const float* __restrict__ X, const float* __restrict__ rsOut,
                             const int* __restrict__ src, const int* __restrict__ dst,
                             float* __restrict__ agg, long long E, int shift)
{
    long long tid = (long long)blockIdx.x * blockDim.x + threadIdx.x;
    long long e = tid >> shift;
    if (e >= E) return;
    int c4 = (int)((tid & ((1LL << shift) - 1)) << 2);
    int K  = 4 << shift;
    int s = src[e], d = dst[e];
    float sc = rsOut[s];
    v4f t = *(const v4f*)(X + (long long)s * K + c4);
    float* o = agg + (long long)d * K + c4;
    atomicAdd(o + 0, t[0] * sc);
    atomicAdd(o + 1, t[1] * sc);
    atomicAdd(o + 2, t[2] * sc);
    atomicAdd(o + 3, t[3] * sc);
}

__global__ void degrees_k(const int* __restrict__ src, const int* __restrict__ dst,
                          float* __restrict__ degO, float* __restrict__ degI, long long E)
{
    long long e = (long long)blockIdx.x * blockDim.x + threadIdx.x;
    if (e >= E) return;
    atomicAdd(degO + src[e], 1.f);
    atomicAdd(degI + dst[e], 1.f);
}

__global__ void to_rsqrt_k(float* __restrict__ d, long long n)
{
    long long i = (long long)blockIdx.x * blockDim.x + threadIdx.x;
    if (i < n) d[i] = rsqrtf(fmaxf(d[i], 1.f));
}

__global__ void zero_f32(float* __restrict__ p, long long n)
{
    long long i = (long long)blockIdx.x * blockDim.x + threadIdx.x;
    if (i < n) p[i] = 0.f;
}

// Convert fp32 weight W[K][Nout] -> transposed f16 Wt[Nout][K]
__global__ void wconv(const float* __restrict__ W, _Float16* __restrict__ Wt, int K, int Nout)
{
    int i = blockIdx.x * blockDim.x + threadIdx.x;
    if (i >= K * Nout) return;
    int n = i / K, k = i % K;
    Wt[i] = (_Float16)W[k * Nout + n];
}

// ---------------------------------------------------------------------------
extern "C" void kernel_launch(void* const* d_in, const int* in_sizes, int n_in,
                              void* d_out, int out_size, void* d_ws, size_t ws_size,
                              hipStream_t stream)
{
    const float* m     = (const float*)d_in[0];
    const float* W1    = (const float*)d_in[1];  const float* b1  = (const float*)d_in[2];
    const float* W2    = (const float*)d_in[3];  const float* b2  = (const float*)d_in[4];
    const float* W3    = (const float*)d_in[5];  const float* b3  = (const float*)d_in[6];
    const float* gamma = (const float*)d_in[7];  const float* beta = (const float*)d_in[8];
    const float* Wg1   = (const float*)d_in[9];  const float* bg1 = (const float*)d_in[10];
    const float* Wg2   = (const float*)d_in[11]; const float* bg2 = (const float*)d_in[12];
    const int*   src   = (const int*)d_in[13];
    const int*   dst   = (const int*)d_in[14];

    const long long E = (long long)in_sizes[0] / 128;   // m is [E,128]
    const long long N = (long long)out_size / 128;      // output is [N,128]

    char* ws = (char*)d_ws;
    size_t off = 0;
    float* h    = (float*)(ws + off); off += (size_t)N * 128 * 4;
    float* agg1 = (float*)(ws + off); off += (size_t)N * 128 * 4;
    float* h2   = (float*)(ws + off); off += (size_t)N * 256 * 4;
    float* degO = (float*)(ws + off); off += (size_t)N * 4;
    float* degI = (float*)(ws + off); off += (size_t)N * 4;
    off = (off + 255) & ~(size_t)255;
    _Float16* W1t  = (_Float16*)(ws + off); off += 128 * 128 * 2;
    _Float16* W2t  = (_Float16*)(ws + off); off += 128 * 128 * 2;
    _Float16* W3t  = (_Float16*)(ws + off); off += 128 * 128 * 2;
    _Float16* Wg1t = (_Float16*)(ws + off); off += 256 * 128 * 2;
    _Float16* Wg2t = (_Float16*)(ws + off); off += 256 * 128 * 2;
    float* agg2 = h;   // alias: [N,256] over h+agg1 (both dead by then)

    const dim3 blk(256);

    // 1) weights -> transposed f16
    wconv<<<(128 * 128 + 255) / 256, blk, 0, stream>>>(W1,  W1t,  128, 128);
    wconv<<<(128 * 128 + 255) / 256, blk, 0, stream>>>(W2,  W2t,  128, 128);
    wconv<<<(128 * 128 + 255) / 256, blk, 0, stream>>>(W3,  W3t,  128, 128);
    wconv<<<(128 * 256 + 255) / 256, blk, 0, stream>>>(Wg1, Wg1t, 128, 256);
    wconv<<<(256 * 128 + 255) / 256, blk, 0, stream>>>(Wg2, Wg2t, 256, 128);

    // 2) zero accumulators + degree buffers (ws is poisoned, not re-zeroed)
    long long zhn = N * 256;                      // h + agg1 contiguous
    zero_f32<<<(unsigned)((zhn + 255) / 256), blk, 0, stream>>>(h, zhn);
    zero_f32<<<(unsigned)((2 * N + 255) / 256), blk, 0, stream>>>(degO, 2 * N);

    // 3) degrees -> rsqrt(max(deg,1))
    degrees_k<<<(unsigned)((E + 255) / 256), blk, 0, stream>>>(src, dst, degO, degI, E);
    to_rsqrt_k<<<(unsigned)((2 * N + 255) / 256), blk, 0, stream>>>(degO, 2 * N);

    // 4) fused edge MLP + LayerNorm + segment_sum(dst) -> h [N,128]
    edge_mlp_scatter<<<(unsigned)((E + 127) / 128), blk, 0, stream>>>(
        m, W1t, b1, W2t, b2, W3t, b3, gamma, beta, dst, h, E);

    // 5) conv1 gather/scatter: agg1[dst] += h[src]*rs_out[src]
    edge_scatter<<<(unsigned)(((E << 5) + 255) / 256), blk, 0, stream>>>(
        h, degO, src, dst, agg1, E, 5);
    // 6) conv1 GEMM: h2 = relu(agg1*rs_in @ Wg1 + bg1)   [N,256]
    node_gemm<128><<<dim3((unsigned)((N + 127) / 128), 2), blk, 0, stream>>>(
        agg1, degI, Wg1t, bg1, h2, N, 256);

    // 7) conv2 scatter into aliased agg2 [N,256]
    zero_f32<<<(unsigned)((zhn + 255) / 256), blk, 0, stream>>>(agg2, zhn);
    edge_scatter<<<(unsigned)(((E << 6) + 255) / 256), blk, 0, stream>>>(
        h2, degO, src, dst, agg2, E, 6);
    // 8) conv2 GEMM -> d_out [N,128]
    node_gemm<256><<<dim3((unsigned)((N + 127) / 128), 1), blk, 0, stream>>>(
        agg2, degI, Wg2t, bg2, (float*)d_out, N, 128);
}
